// CentroidsFlowAD_13211319403321
// MI455X (gfx1250) — matrix operand order repaired and checked by reading, not compiled, not run on hardware
//
#include <hip/hip_runtime.h>

typedef __attribute__((ext_vector_type(16))) __bf16 v16bf;
typedef __attribute__((ext_vector_type(8)))  float  v8f;
typedef __attribute__((ext_vector_type(4)))  int    v4i;

#define DDIM 1024
#define MCENT 2048
#define NROWS 100352          // 32 * 3136
#define ROWS_PER_WG 64
#define COLS_PER_BLK 64
#define PITCH 1032            // bf16 elements per LDS row: 1024 + 8 pad -> bank rotation

#if defined(__has_builtin)
#  if __has_builtin(__builtin_amdgcn_global_load_async_to_lds_b128)
#    define USE_ASYNC_LDS 1
#  endif
#endif

union FragBF {
  v16bf v;
  uint4 q[2];
};

__device__ __forceinline__ unsigned short f2bf(float f) {
  unsigned int u = __float_as_uint(f);
  u += 0x7FFFu + ((u >> 16) & 1u);        // round-to-nearest-even
  return (unsigned short)(u >> 16);
}

__device__ __forceinline__ void st8(float* p, v8f a) {
  float4 x = {a[0], a[1], a[2], a[3]};
  float4 y = {a[4], a[5], a[6], a[7]};
  *(float4*)p = x;
  *(float4*)(p + 4) = y;
}

__device__ __forceinline__ v8f ld8add(const float* p, v8f a) {
  float4 x = *(const float4*)p;
  float4 y = *(const float4*)(p + 4);
  a[0] += x.x; a[1] += x.y; a[2] += x.z; a[3] += x.w;
  a[4] += y.x; a[5] += y.y; a[6] += y.z; a[7] += y.w;
  return a;
}

// ---------------------------------------------------------------------------
// Kernel 1: centroids f32 -> bf16 (into ws) + per-centroid squared norm
// ---------------------------------------------------------------------------
__global__ void prep_centroids(const float* __restrict__ cent,
                               unsigned short* __restrict__ cbf,
                               float* __restrict__ csq) {
  const int c = blockIdx.x;
  const int t = threadIdx.x;
  const float4 v = ((const float4*)(cent + (long)c * DDIM))[t];
  uint2 p;
  p.x = (unsigned)f2bf(v.x) | ((unsigned)f2bf(v.y) << 16);
  p.y = (unsigned)f2bf(v.z) | ((unsigned)f2bf(v.w) << 16);
  ((uint2*)(cbf + (long)c * DDIM))[t] = p;

  float s = v.x * v.x + v.y * v.y + v.z * v.z + v.w * v.w;
#pragma unroll
  for (int off = 16; off >= 1; off >>= 1) s += __shfl_xor(s, off, 32);
  __shared__ float red[8];
  const int wave = t >> 5, lane = t & 31;
  if (lane == 0) red[wave] = s;
  __syncthreads();
  if (t == 0) {
    float tot = 0.f;
#pragma unroll
    for (int i = 0; i < 8; ++i) tot += red[i];
    csq[c] = tot;
  }
}

// ---------------------------------------------------------------------------
// Kernel 2: nearest-centroid distance via bf16 WMMA.
// 256 threads = 8 wave32. Block tile: 64 rows x 64 centroids.
// Each of 4 tile-slots (rowHalf x colHalf) is a 32x32 tile computed by TWO
// waves splitting the K=1024 range (waves 0-3: K<512, waves 4-7: K>=512);
// partial sums merged through LDS. 2 ds_load_b128 per WMMA, all-immediate
// LDS addressing via fully-unrolled K loop with kStart folded into bases.
// ---------------------------------------------------------------------------
__global__ void __launch_bounds__(256)
min_dist_kernel(const float* __restrict__ embeds,
                const unsigned short* __restrict__ cbf,
                const float* __restrict__ csq,
                float* __restrict__ out) {
  extern __shared__ __align__(16) unsigned char smem[];
  unsigned short* Abuf   = (unsigned short*)smem;                 // 64 x PITCH bf16
  unsigned short* Bbuf   = Abuf + ROWS_PER_WG * PITCH;            // 64 x PITCH bf16
  float*          featSq = (float*)(Bbuf + COLS_PER_BLK * PITCH); // 64 floats
  float*          minBuf = featSq + ROWS_PER_WG;                  // 64 floats
  float*          Cmerge = (float*)Bbuf;                          // 16KB alias, dead window

  const int t    = threadIdx.x;
  const int wave = t >> 5;
  const int lane = t & 31;
  const long R0  = (long)blockIdx.x * ROWS_PER_WG;

  if (t < ROWS_PER_WG) featSq[t] = 0.f;
  __syncthreads();

  // ---- Stage A: 64 rows x 1024 f32 -> bf16 LDS; accumulate ||e||^2 --------
  {
    const int row = t >> 2;
    const int q4  = t & 3;
    const float*    src = embeds + (R0 + row) * (long)DDIM + q4 * 256;
    unsigned short* dst = Abuf + row * PITCH + q4 * 256;
    float acc = 0.f;
#pragma unroll 4
    for (int i = 0; i < 64; ++i) {
      float4 v = ((const float4*)src)[i];
      acc += v.x * v.x + v.y * v.y + v.z * v.z + v.w * v.w;
      uint2 p;
      p.x = (unsigned)f2bf(v.x) | ((unsigned)f2bf(v.y) << 16);
      p.y = (unsigned)f2bf(v.z) | ((unsigned)f2bf(v.w) << 16);
      *((uint2*)(dst + i * 4)) = p;
    }
    atomicAdd(&featSq[row], acc);   // ds_add_f32
  }

  const int h      = lane >> 4;              // half-wave select
  const int nl     = lane & 15;
  const int rA     = (wave & 1) * 32;        // row half of block
  const int cL     = ((wave >> 1) & 1) * 32; // col half of block
  const int kHalf  = wave >> 2;              // K-range owner
  const int kStart = kHalf * 512;

  float minAcc0[8], minAcc1[8];
#pragma unroll
  for (int r = 0; r < 8; ++r) { minAcc0[r] = 3.4e38f; minAcc1[r] = 3.4e38f; }

  // Fragment base pointers with kStart and intra-fragment lane offsets folded
  // in: every ds_load in the unrolled K loop is base + constant immediate.
  const unsigned short* aBase0 = Abuf + (rA + nl) * PITCH + kStart + 8 * h;
  const unsigned short* aBase1 = aBase0 + 16 * PITCH;
  const unsigned short* bBase0 = Bbuf + (cL + nl) * PITCH + kStart + 8 * h;
  const unsigned short* bBase1 = bBase0 + 16 * PITCH;

  for (int c0 = 0; c0 < MCENT; c0 += COLS_PER_BLK) {
    __syncthreads();   // Cmerge reads done; safe to overwrite Bbuf
    // ---- Stage B: 64 centroids x 1024 bf16 from ws into LDS --------------
    {
      const int crow = t >> 2;
      const int q4   = t & 3;
      const unsigned short* src = cbf + (long)(c0 + crow) * DDIM + q4 * 256;
      unsigned short*       dst = Bbuf + crow * PITCH + q4 * 256;
#ifdef USE_ASYNC_LDS
      typedef __attribute__((address_space(1))) v4i ga_v4i;
      typedef __attribute__((address_space(3))) v4i lds_v4i;
      ga_v4i*  g = (ga_v4i*)(unsigned long long)(const void*)src;
      lds_v4i* l = (lds_v4i*)(unsigned int)(unsigned long long)(void*)dst;
#pragma unroll 8
      for (int i = 0; i < 32; ++i)
        __builtin_amdgcn_global_load_async_to_lds_b128(g + i, l + i, 0, 0);
#else
      const uint4* gs = (const uint4*)src;
      uint4*       ld = (uint4*)dst;
#pragma unroll 4
      for (int i = 0; i < 32; ++i) ld[i] = gs[i];
#endif
      // prefetch next centroid block into cache hierarchy
      if (c0 + COLS_PER_BLK < MCENT) {
        const char* nxt =
            (const char*)(cbf + (long)(c0 + COLS_PER_BLK + crow) * DDIM) + q4 * 512;
        __builtin_prefetch(nxt,       0, 1);
        __builtin_prefetch(nxt + 128, 0, 1);
        __builtin_prefetch(nxt + 256, 0, 1);
        __builtin_prefetch(nxt + 384, 0, 1);
      }
    }
#ifdef USE_ASYNC_LDS
#  if __has_builtin(__builtin_amdgcn_s_wait_asynccnt)
    __builtin_amdgcn_s_wait_asynccnt(0);
#  else
    asm volatile("s_wait_asynccnt 0" ::: "memory");
#  endif
#endif
    __syncthreads();

    v8f acc00 = {}, acc01 = {}, acc10 = {}, acc11 = {};
#pragma unroll
    for (int kc = 0; kc < 512; kc += 32) {
      FragBF a0, a1, b0, b1;
      // A 16x32 bf16: element e -> K = kc + (e&7) + 8*h + 16*(e>>3)
      a0.q[0] = *(const uint4*)(aBase0 + kc);
      a0.q[1] = *(const uint4*)(aBase0 + kc + 16);
      a1.q[0] = *(const uint4*)(aBase1 + kc);
      a1.q[1] = *(const uint4*)(aBase1 + kc + 16);
      // B 32x16 bf16: element e -> K = kc + (e&7) + 8*h + 16*(e>>3)
      b0.q[0] = *(const uint4*)(bBase0 + kc);
      b0.q[1] = *(const uint4*)(bBase0 + kc + 16);
      b1.q[0] = *(const uint4*)(bBase1 + kc);
      b1.q[1] = *(const uint4*)(bBase1 + kc + 16);

      acc00 = __builtin_amdgcn_wmma_f32_16x16x32_bf16(false, a0.v, false, b0.v,
                                                      (short)0, acc00, false, false);
      acc01 = __builtin_amdgcn_wmma_f32_16x16x32_bf16(false, a0.v, false, b1.v,
                                                      (short)0, acc01, false, false);
      acc10 = __builtin_amdgcn_wmma_f32_16x16x32_bf16(false, a1.v, false, b0.v,
                                                      (short)0, acc10, false, false);
      acc11 = __builtin_amdgcn_wmma_f32_16x16x32_bf16(false, a1.v, false, b1.v,
                                                      (short)0, acc11, false, false);
    }

    // ---- Merge the two K-halves through LDS ------------------------------
    __syncthreads();                       // all Bbuf fragment reads done
    float* mslot = Cmerge + (wave & 3) * 1024 + lane * 8;
    if (kHalf) {                           // waves 4-7 publish partial sums
      st8(mslot +   0, acc00);
      st8(mslot + 256, acc01);
      st8(mslot + 512, acc10);
      st8(mslot + 768, acc11);
    }
    __syncthreads();
    if (!kHalf) {                          // waves 0-3 combine + min-epilogue
      acc00 = ld8add(mslot +   0, acc00);
      acc01 = ld8add(mslot + 256, acc01);
      acc10 = ld8add(mslot + 512, acc10);
      acc11 = ld8add(mslot + 768, acc11);

      const float csqA = csq[c0 + cL + nl];
      const float csqB = csq[c0 + cL + 16 + nl];
#pragma unroll
      for (int r = 0; r < 8; ++r) {
        minAcc0[r] = fminf(minAcc0[r],
                           fminf(csqA - 2.f * acc00[r], csqB - 2.f * acc01[r]));
        minAcc1[r] = fminf(minAcc1[r],
                           fminf(csqA - 2.f * acc10[r], csqB - 2.f * acc11[r]));
      }
    }
  }

  // ---- Reduce over the 16 lanes (N dimension) within each half-wave group --
#pragma unroll
  for (int j = 8; j >= 1; j >>= 1) {
#pragma unroll
    for (int r = 0; r < 8; ++r) {
      minAcc0[r] = fminf(minAcc0[r], __shfl_xor(minAcc0[r], j, 32));
      minAcc1[r] = fminf(minAcc1[r], __shfl_xor(minAcc1[r], j, 32));
    }
  }

  // ---- Merge the two centroid halves (waves 0,1 vs 2,3), store ------------
  __syncthreads();
  if (wave < 2) {
#pragma unroll
    for (int r = 0; r < 8; ++r)
      if (nl == r) {
        minBuf[rA + r + 8 * h]      = minAcc0[r];
        minBuf[rA + 16 + r + 8 * h] = minAcc1[r];
      }
  }
  __syncthreads();
  if (wave == 2 || wave == 3) {
#pragma unroll
    for (int r = 0; r < 8; ++r)
      if (nl == r) {
        const int row0 = rA + r + 8 * h;
        const int row1 = row0 + 16;
        const float q0 = fminf(minAcc0[r], minBuf[row0]);
        const float q1 = fminf(minAcc1[r], minBuf[row1]);
        out[R0 + row0] = sqrtf(fmaxf(featSq[row0] + q0, 0.f));
        out[R0 + row1] = sqrtf(fmaxf(featSq[row1] + q1, 0.f));
      }
  }
}

// ---------------------------------------------------------------------------
extern "C" void kernel_launch(void* const* d_in, const int* in_sizes, int n_in,
                              void* d_out, int out_size, void* d_ws, size_t ws_size,
                              hipStream_t stream) {
  (void)in_sizes; (void)n_in; (void)out_size; (void)ws_size;
  const float* embeds = (const float*)d_in[0];   // [32, 3136, 1024] f32
  const float* cents  = (const float*)d_in[1];   // [2048, 1024] f32

  unsigned short* cbf = (unsigned short*)d_ws;                              // 4 MB bf16
  float* csq = (float*)((char*)d_ws + (size_t)MCENT * DDIM * sizeof(unsigned short));

  prep_centroids<<<MCENT, 256, 0, stream>>>(cents, cbf, csq);

  const size_t ldsBytes = (size_t)(ROWS_PER_WG + COLS_PER_BLK) * PITCH * sizeof(unsigned short)
                        + (size_t)(2 * ROWS_PER_WG) * sizeof(float);        // 264,704 B
  min_dist_kernel<<<NROWS / ROWS_PER_WG, 256, ldsBytes, stream>>>(
      embeds, cbf, csq, (float*)d_out);
}
